// FullAttention_42305427866175
// MI455X (gfx1250) — matrix operand order repaired and checked by reading, not compiled, not run on hardware
//
#include <hip/hip_runtime.h>

// Problem constants (from reference): N=4, L=2048, S=2048, H=8, D=32
#define NB 4
#define LQ 2048
#define SK 2048
#define NH 8
#define DH 32
#define TS 32                 // S tile width per iteration (= WMMA K for the AV GEMM)
#define WAVES 8               // waves per block (wave32) -> 256 threads
#define ROWS_PER_WAVE 16      // query rows per wave (= WMMA M)
#define QBLK (WAVES * ROWS_PER_WAVE)   // 128 query rows per block
#define LDSW 34               // padded row stride (halves) -> conflict-free, 4B aligned pairs

typedef __attribute__((ext_vector_type(16))) _Float16 v16h;
typedef __attribute__((ext_vector_type(8)))  float    v8f;
typedef __attribute__((ext_vector_type(2)))  _Float16 h2;

union Frag16 { v16h h; unsigned int u[8]; };

// pack two f32 -> packed f16 pair (v_cvt_pk_f16_f32 rtz), as our h2 type
__device__ __forceinline__ h2 pk_f16(float a, float b) {
    return __builtin_bit_cast(h2, __builtin_amdgcn_cvt_pkrtz(a, b));
}

// max-reduce across the 16-lane half with fused VOP2+DPP16: one
// v_max_num_f32_dpp per tree level (src0 takes the DPP permute, src1 = x).
// xor1 = quad_perm(1,0,3,2), xor2 = quad_perm(2,3,0,1),
// xor4-equiv = row_half_mirror (lane^7 once 4-groups are uniform),
// xor8-equiv = row_mirror      (lane^15 once 8-groups are uniform).
__device__ __forceinline__ float rowmax16(float x) {
    float t;
    asm("v_max_num_f32_dpp %0, %1, %1 quad_perm:[1,0,3,2] row_mask:0xf bank_mask:0xf bound_ctrl:1"
        : "=v"(t) : "v"(x));
    asm("v_max_num_f32_dpp %0, %1, %1 quad_perm:[2,3,0,1] row_mask:0xf bank_mask:0xf bound_ctrl:1"
        : "=v"(x) : "v"(t));
    asm("v_max_num_f32_dpp %0, %1, %1 row_half_mirror row_mask:0xf bank_mask:0xf bound_ctrl:1"
        : "=v"(t) : "v"(x));
    asm("v_max_num_f32_dpp %0, %1, %1 row_mirror row_mask:0xf bank_mask:0xf bound_ctrl:1"
        : "=v"(x) : "v"(t));
    return x;
}

__global__ __launch_bounds__(WAVES * 32) void
flash_attn_wmma_f16(const float* __restrict__ q,
                    const float* __restrict__ k,
                    const float* __restrict__ v,
                    const unsigned char* __restrict__ qmask,
                    const unsigned char* __restrict__ kvmask,
                    float* __restrict__ out)
{
    __shared__ __align__(16) _Float16 sK [TS * LDSW];                    // [s][d]
    __shared__ __align__(16) _Float16 sVT[DH * LDSW];                    // [d][s]
    __shared__ __align__(16) _Float16 sQ [WAVES][ROWS_PER_WAVE * LDSW];
    __shared__ __align__(16) _Float16 sP [WAVES][ROWS_PER_WAVE * LDSW];

    const int tid   = threadIdx.x;
    const int lane  = tid & 31;
    const int w     = tid >> 5;
    const int ncol  = lane & 15;       // B/C column inside a 16-wide tile
    const int half  = lane >> 4;       // wave half: C rows +8, B K +16
    const int khalf = half << 4;

    const int bid   = blockIdx.x;
    const int qb    = bid & ((LQ / QBLK) - 1);   // 16 q-blocks per (n,h)
    const int nh    = bid >> 4;
    const int n     = nh / NH;
    const int h     = nh % NH;
    const int qrow0 = qb * QBLK + w * ROWS_PER_WAVE;

    // 1/sqrt(D) * log2(e): base-2 softmax so exp is a single v_exp_f32
    const float temp = 0.17677669529663687f * 1.4426950408889634f;

    // ---- stage this wave's 16x32 Q tile (pre-scaled) as f16 ----
    {
        const float* qbase = q + ((size_t)(n * LQ + qrow0) * NH + h) * DH;
        #pragma unroll
        for (int i = 0; i < 4; ++i) {
            int idx = (lane + i * 32) * 4;       // 0..508, 4 contiguous floats each
            int m = idx >> 5, d0 = idx & 31;
            float4 x = *(const float4*)(qbase + (size_t)m * NH * DH + d0);
            *(h2*)&sQ[w][m * LDSW + d0]     = pk_f16(x.x * temp, x.y * temp);
            *(h2*)&sQ[w][m * LDSW + d0 + 2] = pk_f16(x.z * temp, x.w * temp);
        }
    }
    __syncthreads();

    // ---- A-fragment of Q (16x32 f16 A-matrix layout), kept in VGPRs ----
    Frag16 aq;
    #pragma unroll
    for (int j = 0; j < 8; ++j) {
        int kq = khalf + 2 * j;
        aq.u[j] = *(const unsigned int*)&sQ[w][ncol * LDSW + kq];
    }

    // constant all-ones B fragment: row-sum of P via WMMA (P x 1 = row sums)
    Frag16 bones;
    #pragma unroll
    for (int j = 0; j < 8; ++j) bones.u[j] = 0x3C003C00u;   // packed f16 1.0,1.0

    v8f acc0 = {0.f,0.f,0.f,0.f,0.f,0.f,0.f,0.f};   // O cols d=0..15
    v8f acc1 = {0.f,0.f,0.f,0.f,0.f,0.f,0.f,0.f};   // O cols d=16..31
    float rmax[8], rsum[8];
    #pragma unroll
    for (int j = 0; j < 8; ++j) { rmax[j] = -1e30f; rsum[j] = 0.f; }

    const unsigned char* kvm = kvmask + (size_t)n * SK;
    const float* kb0 = k + ((size_t)(n * SK) * NH + h) * DH;
    const float* vb0 = v + ((size_t)(n * SK) * NH + h) * DH;

    // this thread's slice of the cooperative 32x32 staging: 4 contiguous floats
    const int st  = tid >> 3;            // source row s = 0..31
    const int dt  = (tid & 7) << 2;      // source col d = 0,4,...,28

    for (int s0 = 0; s0 < SK; s0 += TS) {
        __syncthreads();   // all waves done reading previous K/V tiles
        {
            size_t g = (size_t)(s0 + st) * NH * DH + dt;
            float4 kx = *(const float4*)(kb0 + g);
            float4 vx = *(const float4*)(vb0 + g);
            // K tile: packed 4-byte DS stores (d contiguous)
            *(h2*)&sK[st * LDSW + dt]     = pk_f16(kx.x, kx.y);
            *(h2*)&sK[st * LDSW + dt + 2] = pk_f16(kx.z, kx.w);
            // V tile transposed: 2-byte scatter (17-dword row stride: conflict-free)
            sVT[(dt + 0) * LDSW + st] = (_Float16)vx.x;
            sVT[(dt + 1) * LDSW + st] = (_Float16)vx.y;
            sVT[(dt + 2) * LDSW + st] = (_Float16)vx.z;
            sVT[(dt + 3) * LDSW + st] = (_Float16)vx.w;
        }
        __syncthreads();

        // prefetch next S-tile of K/V into cache while we compute this one
        if (s0 + TS < SK) {
            size_t gn = (size_t)(s0 + TS + (tid & 31)) * NH * DH;
            __builtin_prefetch(kb0 + gn, 0, 3);
            __builtin_prefetch(vb0 + gn, 0, 3);
        }

        // ---- fragment loads from LDS (B of K^T and B of V; hoisted) ----
        Frag16 bk0, bk1, bv0, bv1;
        #pragma unroll
        for (int j = 0; j < 8; ++j) {
            bk0.u[j] = *(const unsigned int*)&sK [(ncol     ) * LDSW + khalf + 2 * j];
            bk1.u[j] = *(const unsigned int*)&sK [(ncol + 16) * LDSW + khalf + 2 * j];
            bv0.u[j] = *(const unsigned int*)&sVT[(ncol     ) * LDSW + khalf + 2 * j];
            bv1.u[j] = *(const unsigned int*)&sVT[(ncol + 16) * LDSW + khalf + 2 * j];
        }

        v8f zero = {0.f,0.f,0.f,0.f,0.f,0.f,0.f,0.f};
        v8f c0 = __builtin_amdgcn_wmma_f32_16x16x32_f16(false, aq.h, false, bk0.h,
                                                        (short)0, zero, false, false);
        v8f c1 = __builtin_amdgcn_wmma_f32_16x16x32_f16(false, aq.h, false, bk1.h,
                                                        (short)0, zero, false, false);

        const float mcol0 = kvm[s0 + ncol]      ? 0.f : -1e30f;
        const float mcol1 = kvm[s0 + 16 + ncol] ? 0.f : -1e30f;

        // ---- online softmax (base 2): row max via fused DPP, row sum via WMMA ----
        float tmax[8];
        #pragma unroll
        for (int j = 0; j < 8; ++j) {
            c0[j] += mcol0;
            c1[j] += mcol1;
            tmax[j] = rowmax16(fmaxf(c0[j], c1[j]));
        }
        float alpha[8];
        #pragma unroll
        for (int j = 0; j < 8; ++j) {
            float mnew = fmaxf(rmax[j], tmax[j]);
            alpha[j]   = __builtin_amdgcn_exp2f(rmax[j] - mnew);
            rmax[j]    = mnew;
            float p0 = __builtin_amdgcn_exp2f(c0[j] - mnew);
            float p1 = __builtin_amdgcn_exp2f(c1[j] - mnew);
            acc0[j] *= alpha[j];
            acc1[j] *= alpha[j];
            int row = j + (half << 3);
            sP[w][row * LDSW + ncol]      = (_Float16)p0;   // cols s0..s0+15
            sP[w][row * LDSW + 16 + ncol] = (_Float16)p1;   // cols s0+16..s0+31
        }
        // same-wave DS ops execute in order; block compiler reordering only.
        __builtin_amdgcn_wave_barrier();

        // ---- re-layout P (C-format -> A-format) from per-wave LDS scratch ----
        Frag16 ap;
        #pragma unroll
        for (int j = 0; j < 8; ++j) {
            int kp = khalf + 2 * j;
            ap.u[j] = *(const unsigned int*)&sP[w][ncol * LDSW + kp];
        }

        // row sums of P ride on a WMMA against the all-ones B matrix
        v8f ps = __builtin_amdgcn_wmma_f32_16x16x32_f16(false, ap.h, false, bones.h,
                                                        (short)0, zero, false, false);
        acc0 = __builtin_amdgcn_wmma_f32_16x16x32_f16(false, ap.h, false, bv0.h,
                                                      (short)0, acc0, false, false);
        acc1 = __builtin_amdgcn_wmma_f32_16x16x32_f16(false, ap.h, false, bv1.h,
                                                      (short)0, acc1, false, false);
        #pragma unroll
        for (int j = 0; j < 8; ++j)
            rsum[j] = rsum[j] * alpha[j] + ps[j];
    }

    // ---- epilogue: normalize by row sum, apply q_mask, coalesced f32 stores ----
    #pragma unroll
    for (int j = 0; j < 8; ++j) {
        int row = j + (half << 3);
        int gl  = qrow0 + row;
        float qm  = qmask[(size_t)n * LQ + gl] ? 1.f : 0.f;
        float inv = (rsum[j] > 0.f) ? (qm / rsum[j]) : 0.f;
        size_t ob = ((size_t)(n * LQ + gl) * NH + h) * DH;
        out[ob + ncol]      = acc0[j] * inv;
        out[ob + 16 + ncol] = acc1[j] * inv;
    }
}

extern "C" void kernel_launch(void* const* d_in, const int* in_sizes, int n_in,
                              void* d_out, int out_size, void* d_ws, size_t ws_size,
                              hipStream_t stream) {
    (void)in_sizes; (void)n_in; (void)out_size; (void)d_ws; (void)ws_size;
    const float*         q   = (const float*)d_in[0];
    const float*         k   = (const float*)d_in[1];
    const float*         v   = (const float*)d_in[2];
    const unsigned char* qm  = (const unsigned char*)d_in[3];
    const unsigned char* kvm = (const unsigned char*)d_in[4];
    float* out = (float*)d_out;

    dim3 grid(NB * NH * (LQ / QBLK));   // 4*8*16 = 512 blocks
    dim3 block(WAVES * 32);             // 256 threads = 8 wave32
    flash_attn_wmma_f16<<<grid, block, 0, stream>>>(q, k, v, qm, kvm, out);
}
